// VectorQuantizer_7267084665645
// MI455X (gfx1250) — matrix operand order repaired and checked by reading, not compiled, not run on hardware
//
#include <hip/hip_runtime.h>
#include <hip/hip_bf16.h>

typedef __bf16 bf16_t;
typedef __attribute__((ext_vector_type(16))) __bf16 v16bf;
typedef __attribute__((ext_vector_type(8)))  float  v8f;
typedef int v4i_vs __attribute__((vector_size(16)));   // 4 x i32, matches builtin proto

#define DECAY_  0.999f
#define EPS_    1e-5f
#define COMMIT_ 0.25f

static constexpr int Bb = 8, Tt = 4096, Dd = 512, Mm = 1024, Hh = 512;
static constexpr int Nn = Bb * Tt;          // 32768 tokens
static constexpr int KT = Dd / 32;          // 16 K-tiles of 32 (K=512 everywhere)
static constexpr int BSTRIDE = Dd + 8;      // padded LDS column stride (bank spread)

#ifndef __has_builtin
#define __has_builtin(x) 0
#endif
#if __has_builtin(__builtin_amdgcn_global_load_async_to_lds_b128)
#define HAVE_ASYNC_LDS 1
#else
#define HAVE_ASYNC_LDS 0
#endif
#if __has_builtin(__builtin_amdgcn_s_wait_asynccnt)
#define HAVE_WAIT_ASYNC 1
#else
#define HAVE_WAIT_ASYNC 0
#endif

#if HAVE_ASYNC_LDS
#if HAVE_WAIT_ASYNC
#define ASYNC_WAIT(n) __builtin_amdgcn_s_wait_asynccnt(n)
#else
#define ASYNC_WAIT(n) asm volatile("s_wait_asynccnt " #n ::: "memory")
#endif
#else
#define ASYNC_WAIT(n) ((void)0)
#endif

// ---------------- fragment loaders (wave32 WMMA layouts) ----------------

// A-matrix 16x32 bf16 fragment from an f32 row. p points at row + kt*32 + kb,
// kb = (lane>>4)*8. Elems 0..7 = K(+0..7), elems 8..15 = K(+16..23).
__device__ inline v16bf frag_a_from_f32(const float* __restrict__ p) {
  v16bf f;
#pragma unroll
  for (int j = 0; j < 8; ++j) f[j] = (bf16_t)p[j];
#pragma unroll
  for (int j = 0; j < 8; ++j) f[8 + j] = (bf16_t)p[16 + j];
  return f;
}

// 16 contiguous bf16 (32B) -> fragment (B columns; LDS or global source).
__device__ inline v16bf frag_from_bf16(const bf16_t* __restrict__ p) {
  union { v16bf v; uint4 u[2]; } r;
  r.u[0] = *(const uint4*)(p);
  r.u[1] = *(const uint4*)(p + 8);
  return r.v;
}

// A fragment from a bf16 row: two 16B loads (K+0..7 and K+16..23 halves).
__device__ inline v16bf frag_a_from_bf16(const bf16_t* __restrict__ p) {
  union { v16bf v; uint4 u[2]; } r;
  r.u[0] = *(const uint4*)(p);
  r.u[1] = *(const uint4*)(p + 16);
  return r.v;
}

// ---------------- async global->LDS staging of one 16x512 B tile --------

__device__ inline void lds_cp16(const bf16_t* __restrict__ g, bf16_t* l) {
#if HAVE_ASYNC_LDS
  __builtin_amdgcn_global_load_async_to_lds_b128(
      (__attribute__((address_space(1))) v4i_vs*)g,
      (__attribute__((address_space(3))) v4i_vs*)l, 0, 0);
#else
  *(uint4*)l = *(const uint4*)g;
#endif
}

// Tile = 16 consecutive rows of Bt (contiguous 16KB); pad columns in LDS.
__device__ inline void stage_tile(const bf16_t* __restrict__ gBase,
                                  bf16_t* sDst, int tid) {
#pragma unroll
  for (int j = 0; j < 4; ++j) {
    const int o = tid + j * 256;      // 0..1023 ops of 16B
    const int c = o >> 6;             // column 0..15
    const int w = (o & 63) << 3;      // elem offset within column
    lds_cp16(gBase + c * Dd + w, sDst + c * BSTRIDE + w);
  }
}

// ---------------- shared GEMM sweep: LDS-staged B, double buffered ------

template <typename F>
__device__ inline void gemm_sweep(const bf16_t* __restrict__ Bt,
                                  const v16bf (&a)[KT],
                                  bf16_t (*sB)[16 * BSTRIDE],
                                  int NT, F&& per_tile) {
  const int tid  = threadIdx.x;
  const int lane = tid & 31;
  const int kh   = (lane >> 4) * 8;
  const int myc  = lane & 15;

  stage_tile(Bt, sB[0], tid);
  for (int nt = 0; nt < NT; ++nt) {
    const int cur = nt & 1;
    if (nt + 1 < NT) {
      stage_tile(Bt + (size_t)(nt + 1) * 16 * Dd, sB[cur ^ 1], tid);
      ASYNC_WAIT(4);                  // current tile's 4 ops have landed
    } else {
      ASYNC_WAIT(0);
    }
    __syncthreads();                  // all waves' staging visible
    const bf16_t* bcol = sB[cur] + myc * BSTRIDE + kh;
    v8f acc = {};
#pragma unroll
    for (int kt = 0; kt < KT; ++kt) {
      v16bf b = frag_a_from_bf16(bcol + kt * 32);   // ds_load_b128 x2
      acc = __builtin_amdgcn_wmma_f32_16x16x32_bf16(false, a[kt], false, b,
                                                    (short)0, acc, false, false);
    }
    per_tile(nt, acc);
    __syncthreads();                  // done reading before next overwrite
  }
}

// ---------------- prep: bf16 codebook, transposed bf16 weights, ||e||^2 --

__global__ void k_prep(const float* __restrict__ emb, const float* __restrict__ W1,
                       const float* __restrict__ W2, bf16_t* __restrict__ embb,
                       bf16_t* __restrict__ W1t, bf16_t* __restrict__ W2t,
                       float* __restrict__ esq) {
  const int id = blockIdx.x * blockDim.x + threadIdx.x;
  const int stride = gridDim.x * blockDim.x;
  for (int i = id; i < Mm * Dd; i += stride) embb[i] = (bf16_t)emb[i];
  for (int i = id; i < Hh * Dd; i += stride) {          // W1t[n][k] = W1[k][n]
    int n = i / Dd, k = i % Dd;
    W1t[i] = (bf16_t)W1[(size_t)k * Hh + n];
  }
  for (int i = id; i < Mm * Hh; i += stride) {          // W2t[n][k] = W2[k][n]
    int n = i / Hh, k = i % Hh;
    W2t[i] = (bf16_t)W2[(size_t)k * Mm + n];
  }
  if (id < Mm) {
    float s = 0.f;
    const float* p = emb + (size_t)id * Dd;
    for (int d = 0; d < Dd; ++d) s += p[d] * p[d];
    esq[id] = s;
  }
}

// ---------------- distances + argmin via bf16 WMMA ----------------------

__global__ __launch_bounds__(256) void k_argmin(
    const float* __restrict__ x, const bf16_t* __restrict__ embb,
    const float* __restrict__ esq, int* __restrict__ indices) {
  __shared__ bf16_t sB[2][16 * BSTRIDE];
  const int lane = threadIdx.x & 31;
  const int wave = threadIdx.x >> 5;
  const int row0 = blockIdx.x * 128 + wave * 16;
  const int r    = lane & 15;
  const int kb   = (lane >> 4) * 8;
  const int myc  = lane & 15;

  const float* xrow = x + (size_t)(row0 + r) * Dd + kb;
  v16bf a[KT];
#pragma unroll
  for (int kt = 0; kt < KT; ++kt) a[kt] = frag_a_from_f32(xrow + kt * 32);

  float minv[8]; int mini[8];
#pragma unroll
  for (int i = 0; i < 8; ++i) { minv[i] = 3.4e38f; mini[i] = 0; }

  gemm_sweep(embb, a, sB, Mm / 16, [&](int nt, const v8f& acc) {
    const int n0 = nt * 16;
    const float e2 = esq[n0 + myc];
#pragma unroll
    for (int i = 0; i < 8; ++i) {
      float s = e2 - 2.0f * acc[i];
      if (s < minv[i]) { minv[i] = s; mini[i] = n0 + myc; }
    }
  });

  // reduce (min, argmin) across the 16-lane row groups (halves in parallel)
#pragma unroll
  for (int m = 1; m < 16; m <<= 1) {
#pragma unroll
    for (int i = 0; i < 8; ++i) {
      float ov = __shfl_xor(minv[i], m, 32);
      int   oi = __shfl_xor(mini[i], m, 32);
      if (ov < minv[i]) { minv[i] = ov; mini[i] = oi; }
    }
  }
  if ((lane & 15) == 0) {
#pragma unroll
    for (int i = 0; i < 8; ++i)
      indices[row0 + i + ((lane >= 16) ? 8 : 0)] = mini[i];
  }
}

// ---------------- gather/quantize + commit loss + scatter adds ----------

__global__ __launch_bounds__(256) void k_quant(
    const float* __restrict__ x, const float* __restrict__ emb,
    const int* __restrict__ indices, float* __restrict__ out_q,
    float* __restrict__ vqp, float* __restrict__ counts, float* __restrict__ dw) {
  const int t = blockIdx.x;
  const int idx = indices[t];
  const float* xr = x + (size_t)t * Dd;
  const float* er = emb + (size_t)idx * Dd;
  float* dwr = dw + (size_t)idx * Dd;
  float local = 0.f;
#pragma unroll
  for (int j = 0; j < 2; ++j) {
    const int d = threadIdx.x + j * 256;
    const float xv = xr[d], ev = er[d];
    out_q[(size_t)t * Dd + d] = xv + (ev - xv);   // straight-through quantized
    const float diff = xv - ev;
    local += diff * diff;
    atomicAdd(&dwr[d], xv);                       // encodings^T @ x scatter
  }
  __shared__ float sd[256];
  sd[threadIdx.x] = local;
  __syncthreads();
  for (int s = 128; s > 0; s >>= 1) {
    if (threadIdx.x < s) sd[threadIdx.x] += sd[threadIdx.x + s];
    __syncthreads();
  }
  if (threadIdx.x == 0) {
    vqp[t] = COMMIT_ * sd[0];
    atomicAdd(&counts[idx], 1.0f);
  }
}

// ---------------- EMA count normalize + perplexity (single block) -------

__global__ void k_ema1(const float* __restrict__ counts,
                       const float* __restrict__ ema_count,
                       float* __restrict__ out_newcount, float* __restrict__ out_perp) {
  __shared__ float sd[1024];
  const int m = threadIdx.x;
  const float c = counts[m];
  const float ncr = DECAY_ * ema_count[m] + (1.0f - DECAY_) * c;
  sd[m] = ncr; __syncthreads();
  for (int s = 512; s > 0; s >>= 1) { if (m < s) sd[m] += sd[m + s]; __syncthreads(); }
  const float n = sd[0]; __syncthreads();
  out_newcount[m] = (ncr + EPS_) / (n + (float)Mm * EPS_) * n;
  const float avg = c / (float)Nn;
  sd[m] = avg * logf(avg + 1e-10f); __syncthreads();
  for (int s = 512; s > 0; s >>= 1) { if (m < s) sd[m] += sd[m + s]; __syncthreads(); }
  if (m == 0) out_perp[0] = expf(-sd[0]);
}

__global__ void k_ema2(const float* __restrict__ ema_weight, const float* __restrict__ dw,
                       const float* __restrict__ newcount,
                       float* __restrict__ out_nw, float* __restrict__ out_nemb) {
  const int id = blockIdx.x * 256 + threadIdx.x;   // Mm*Dd threads
  const float nw = DECAY_ * ema_weight[id] + (1.0f - DECAY_) * dw[id];
  out_nw[id] = nw;
  out_nemb[id] = nw / newcount[id / Dd];
}

// ---------------- LayerNorm + ReLU -> bf16 (f32 and bf16 sources) -------

__global__ __launch_bounds__(256) void k_ln_relu_f32(
    const float* __restrict__ x, const float* __restrict__ g,
    const float* __restrict__ b, bf16_t* __restrict__ out) {
  const int t = blockIdx.x;
  const float* xr = x + (size_t)t * Dd;
  const float v0 = xr[threadIdx.x], v1 = xr[threadIdx.x + 256];
  __shared__ float sd[256];
  sd[threadIdx.x] = v0 + v1; __syncthreads();
  for (int s = 128; s > 0; s >>= 1) { if (threadIdx.x < s) sd[threadIdx.x] += sd[threadIdx.x + s]; __syncthreads(); }
  const float mu = sd[0] / (float)Dd; __syncthreads();
  const float d0 = v0 - mu, d1 = v1 - mu;
  sd[threadIdx.x] = d0 * d0 + d1 * d1; __syncthreads();
  for (int s = 128; s > 0; s >>= 1) { if (threadIdx.x < s) sd[threadIdx.x] += sd[threadIdx.x + s]; __syncthreads(); }
  const float rsv = rsqrtf(sd[0] / (float)Dd + 1e-5f);
  const float y0 = fmaxf(d0 * rsv * g[threadIdx.x] + b[threadIdx.x], 0.f);
  const float y1 = fmaxf(d1 * rsv * g[threadIdx.x + 256] + b[threadIdx.x + 256], 0.f);
  out[(size_t)t * Dd + threadIdx.x]       = (bf16_t)y0;
  out[(size_t)t * Dd + threadIdx.x + 256] = (bf16_t)y1;
}

__global__ __launch_bounds__(256) void k_ln_relu_bf16(
    const bf16_t* __restrict__ x, const float* __restrict__ g,
    const float* __restrict__ b, bf16_t* __restrict__ out) {
  const int t = blockIdx.x;
  const bf16_t* xr = x + (size_t)t * Hh;
  const float v0 = (float)xr[threadIdx.x], v1 = (float)xr[threadIdx.x + 256];
  __shared__ float sd[256];
  sd[threadIdx.x] = v0 + v1; __syncthreads();
  for (int s = 128; s > 0; s >>= 1) { if (threadIdx.x < s) sd[threadIdx.x] += sd[threadIdx.x + s]; __syncthreads(); }
  const float mu = sd[0] / (float)Hh; __syncthreads();
  const float d0 = v0 - mu, d1 = v1 - mu;
  sd[threadIdx.x] = d0 * d0 + d1 * d1; __syncthreads();
  for (int s = 128; s > 0; s >>= 1) { if (threadIdx.x < s) sd[threadIdx.x] += sd[threadIdx.x + s]; __syncthreads(); }
  const float rsv = rsqrtf(sd[0] / (float)Hh + 1e-5f);
  const float y0 = fmaxf(d0 * rsv * g[threadIdx.x] + b[threadIdx.x], 0.f);
  const float y1 = fmaxf(d1 * rsv * g[threadIdx.x + 256] + b[threadIdx.x + 256], 0.f);
  out[(size_t)t * Hh + threadIdx.x]       = (bf16_t)y0;
  out[(size_t)t * Hh + threadIdx.x + 256] = (bf16_t)y1;
}

// ---------------- bf16 WMMA GEMM: Out[rows,Ncols] = A[rows,512] @ Bt^T ---

__global__ __launch_bounds__(256) void k_gemm_bf16(
    const bf16_t* __restrict__ A, const bf16_t* __restrict__ Bt,
    bf16_t* __restrict__ Out, int Ncols) {
  __shared__ bf16_t sB[2][16 * BSTRIDE];
  const int lane = threadIdx.x & 31;
  const int wave = threadIdx.x >> 5;
  const int row0 = blockIdx.x * 128 + wave * 16;
  const int r    = lane & 15;
  const int kb   = (lane >> 4) * 8;
  const int myc  = lane & 15;

  const bf16_t* arow = A + (size_t)(row0 + r) * Dd + kb;
  v16bf a[KT];
#pragma unroll
  for (int kt = 0; kt < KT; ++kt) a[kt] = frag_a_from_bf16(arow + kt * 32);

  gemm_sweep(Bt, a, sB, Ncols / 16, [&](int nt, const v8f& acc) {
    const int col = nt * 16 + myc;
#pragma unroll
    for (int i = 0; i < 8; ++i) {
      const int row = row0 + i + ((lane >= 16) ? 8 : 0);
      Out[(size_t)row * Ncols + col] = (bf16_t)acc[i];
    }
  });
}

// ------ final GEMM fused with online log-softmax + cross-entropy --------

__global__ __launch_bounds__(256) void k_gemm_ce(
    const bf16_t* __restrict__ A /* g2 [N,H] */, const bf16_t* __restrict__ Bt /* W2t [M,H] */,
    const int* __restrict__ labels, const int* __restrict__ mask,
    const float* __restrict__ vqp, float* __restrict__ out_loss) {
  __shared__ bf16_t sB[2][16 * BSTRIDE];
  const int lane = threadIdx.x & 31;
  const int wave = threadIdx.x >> 5;
  const int row0 = blockIdx.x * 128 + wave * 16;
  const int r    = lane & 15;
  const int kb   = (lane >> 4) * 8;
  const int myc  = lane & 15;

  const bf16_t* arow = A + (size_t)(row0 + r) * Hh + kb;
  v16bf a[KT];
#pragma unroll
  for (int kt = 0; kt < KT; ++kt) a[kt] = frag_a_from_bf16(arow + kt * 32);

  float rm[8], rs[8], lv[8]; int lab[8], fnd[8];
#pragma unroll
  for (int i = 0; i < 8; ++i) {
    const int row = row0 + i + ((lane >= 16) ? 8 : 0);
    lab[i] = mask[row] ? -1 : labels[row];
    rm[i] = -3.4e38f; rs[i] = 0.f; lv[i] = 0.f; fnd[i] = 0;
  }

  gemm_sweep(Bt, a, sB, Mm / 16, [&](int nt, const v8f& acc) {
    const int col = nt * 16 + myc;
#pragma unroll
    for (int i = 0; i < 8; ++i) {               // online softmax update
      const float lg = acc[i];
      const float mn = fmaxf(rm[i], lg);
      rs[i] = rs[i] * __expf(rm[i] - mn) + __expf(lg - mn);
      rm[i] = mn;
      if (col == lab[i]) { lv[i] = lg; fnd[i] = 1; }
    }
  });

  // merge softmax state across 16-lane row groups
#pragma unroll
  for (int m = 1; m < 16; m <<= 1) {
#pragma unroll
    for (int i = 0; i < 8; ++i) {
      const float om = __shfl_xor(rm[i], m, 32);
      const float os = __shfl_xor(rs[i], m, 32);
      const float mn = fmaxf(rm[i], om);
      rs[i] = rs[i] * __expf(rm[i] - mn) + os * __expf(om - mn);
      rm[i] = mn;
      const float ov = __shfl_xor(lv[i], m, 32);
      const int   of = __shfl_xor(fnd[i], m, 32);
      if (of && !fnd[i]) { lv[i] = ov; fnd[i] = 1; }
    }
  }
  if ((lane & 15) == 0) {
#pragma unroll
    for (int i = 0; i < 8; ++i) {
      const int row = row0 + i + ((lane >= 16) ? 8 : 0);
      const float ce = (lab[i] >= 0) ? -(lv[i] - rm[i] - logf(rs[i])) : 0.0f;
      out_loss[row] = ce + vqp[row];
    }
  }
}

// ------------------------------- launch ---------------------------------

extern "C" void kernel_launch(void* const* d_in, const int* in_sizes, int n_in,
                              void* d_out, int out_size, void* d_ws, size_t ws_size,
                              hipStream_t stream) {
  (void)in_sizes; (void)n_in; (void)out_size; (void)ws_size;
  const float* x        = (const float*)d_in[0];
  const int*   mask     = (const int*)d_in[1];
  const int*   labels   = (const int*)d_in[2];
  const float* emb      = (const float*)d_in[3];
  const float* ema_cnt  = (const float*)d_in[4];
  const float* ema_wt   = (const float*)d_in[5];
  const float* ln1_g    = (const float*)d_in[6];
  const float* ln1_b    = (const float*)d_in[7];
  const float* W1       = (const float*)d_in[8];
  const float* ln2_g    = (const float*)d_in[9];
  const float* ln2_b    = (const float*)d_in[10];
  const float* W2       = (const float*)d_in[11];

  float* out      = (float*)d_out;
  float* out_q    = out;                                 // [N, D]
  float* out_loss = out + (size_t)Nn * Dd;               // [N]
  float* out_perp = out_loss + Nn;                       // [1]
  float* out_nemb = out_perp + 1;                        // [M, D]
  float* out_ncnt = out_nemb + (size_t)Mm * Dd;          // [M]
  float* out_nw   = out_ncnt + Mm;                       // [M, D]

  char* ws = (char*)d_ws;
  bf16_t* embb  = (bf16_t*)(ws + 0);                         // 1 MiB
  bf16_t* W1t   = (bf16_t*)(ws + 1048576);                   // 512 KiB
  bf16_t* W2t   = (bf16_t*)(ws + 1572864);                   // 1 MiB
  float*  esq   = (float*) (ws + 2621440);                   // 4 KiB
  float*  counts= (float*) (ws + 2625536);                   // 4 KiB
  int*    idxs  = (int*)   (ws + 2629632);                   // 128 KiB
  float*  vqp   = (float*) (ws + 2760704);                   // 128 KiB
  float*  dw    = (float*) (ws + 2891776);                   // 2 MiB
  bf16_t* g1    = (bf16_t*)(ws + 4988928);                   // 32 MiB
  bf16_t* hbf   = (bf16_t*)(ws + 38543360);                  // 32 MiB
  bf16_t* g2    = (bf16_t*)(ws + 72097792);                  // 32 MiB

  (void)hipMemsetAsync(counts, 0, Mm * sizeof(float), stream);
  (void)hipMemsetAsync(dw, 0, (size_t)Mm * Dd * sizeof(float), stream);

  k_prep<<<512, 256, 0, stream>>>(emb, W1, W2, embb, W1t, W2t, esq);
  k_argmin<<<Nn / 128, 256, 0, stream>>>(x, embb, esq, idxs);
  k_quant<<<Nn, 256, 0, stream>>>(x, emb, idxs, out_q, vqp, counts, dw);
  k_ema1<<<1, 1024, 0, stream>>>(counts, ema_cnt, out_ncnt, out_perp);
  k_ema2<<<(Mm * Dd) / 256, 256, 0, stream>>>(ema_wt, dw, out_ncnt, out_nw, out_nemb);
  k_ln_relu_f32<<<Nn, 256, 0, stream>>>(x, ln1_g, ln1_b, g1);
  k_gemm_bf16<<<Nn / 128, 256, 0, stream>>>(g1, W1t, hbf, Hh);
  k_ln_relu_bf16<<<Nn, 256, 0, stream>>>(hbf, ln2_g, ln2_b, g2);
  k_gemm_ce<<<Nn / 128, 256, 0, stream>>>(g2, W2t, labels, mask, vqp, out_loss);
}
